// FoodRiskGNN_18219251270415
// MI455X (gfx1250) — matrix operand order repaired and checked by reading, not compiled, not run on hardware
//
#include <hip/hip_runtime.h>
#include <hip/hip_bf16.h>
#include <math.h>

typedef __attribute__((ext_vector_type(16))) __bf16         v16bf;
typedef __attribute__((ext_vector_type(16))) unsigned short v16u;
typedef __attribute__((ext_vector_type(8)))  unsigned short v8us;
typedef __attribute__((ext_vector_type(8)))  unsigned int   v8u;
typedef __attribute__((ext_vector_type(8)))  float          v8f;

// round-to-nearest-even f32 -> bf16 (bit trick)
__device__ __forceinline__ unsigned short f2bf(float f) {
  unsigned int u = __builtin_bit_cast(unsigned int, f);
  u += 0x7FFFu + ((u >> 16) & 1u);
  return (unsigned short)(u >> 16);
}

// ---------------- degree / scatter kernels (bandwidth+atomic bound) ----------------

__global__ void k_degree(const int* __restrict__ dst, float* __restrict__ deg, int E) {
  int e = blockIdx.x * blockDim.x + threadIdx.x;
  if (e < E) atomicAdd(&deg[dst[e]], 1.0f);
}

__global__ void k_invdeg(float* __restrict__ deg, int N) {
  int i = blockIdx.x * blockDim.x + threadIdx.x;
  if (i < N) deg[i] = 1.0f / fmaxf(deg[i], 1.0f);
}

// thread t handles edge e = t / (C/4), channels [4c, 4c+4): coalesced float4 gather,
// 4x global_atomic_add_f32 scatter (working set is L2-resident: 192MB L2).
template<int C>
__global__ void k_scatter(const float* __restrict__ feat, const int* __restrict__ src,
                          const int* __restrict__ dst, float* __restrict__ agg, int E) {
  constexpr int CH = C / 4;
  long long t = (long long)blockIdx.x * blockDim.x + threadIdx.x;
  if (t >= (long long)E * CH) return;
  int e = (int)(t / CH);
  int c = (int)(t % CH) * 4;
  int s = src[e], d = dst[e];
  const float4 v = *(const float4*)(feat + (size_t)s * C + c);
  float* p = agg + (size_t)d * C + c;
  atomicAdd(p + 0, v.x);
  atomicAdd(p + 1, v.y);
  atomicAdd(p + 2, v.z);
  atomicAdd(p + 3, v.w);
}

// ---------------- f32 -> bf16 activation convert (optional per-row invdeg scale) ----------
// Each thread: one float4 -> 4 bf16 (8B store). Converts each element exactly once,
// instead of 8x (per wave) inside the GEMM.
template<int C, bool SCALE>
__global__ void k_cvt_bf(const float* __restrict__ in, const float* __restrict__ invdeg,
                         unsigned short* __restrict__ outp, int N) {
  constexpr int CH = C / 4;
  long long t = (long long)blockIdx.x * blockDim.x + threadIdx.x;
  if (t >= (long long)N * CH) return;
  float4 v = ((const float4*)in)[t];
  if (SCALE) {
    const float s = invdeg[(int)(t / CH)];
    v.x *= s; v.y *= s; v.z *= s; v.w *= s;
  }
  unsigned int lo = (unsigned int)f2bf(v.x) | ((unsigned int)f2bf(v.y) << 16);
  unsigned int hw = (unsigned int)f2bf(v.z) | ((unsigned int)f2bf(v.w) << 16);
  ((uint2*)outp)[t] = make_uint2(lo, hw);
}

// ---------------- weight pre-pack into WMMA B-fragment order ----------------
// B 32x16 bf16 fragment (ISA 7.12.2, wave32): lane L -> n=L&15, hi=L>>4; elem i -> k=8*hi+i (i<8), k=16+8*hi+(i-8) (i>=8).
// Packed layout: P[((kb*CT + ct)*32 + lane)*16 + i], CT = NCOL/16.
template<int K, int NCOL>
__global__ void k_pack_w(const float* __restrict__ W, unsigned short* __restrict__ P) {
  constexpr int CT  = NCOL / 16;
  constexpr int TOT = K * NCOL;
  int t = blockIdx.x * blockDim.x + threadIdx.x;
  if (t >= TOT) return;
  int i    = t & 15;
  int lane = (t >> 4) & 31;
  int ct   = (t >> 9) % CT;
  int kb   = t / (512 * CT);
  int hiL  = lane >> 4;
  int n    = lane & 15;
  int k    = kb * 32 + (i < 8 ? 8 * hiL + i : 16 + 8 * hiL + (i - 8));
  int col  = ct * 16 + n;
  P[t] = f2bf(W[(size_t)k * NCOL + col]);
}

// ---------------- fused SAGE GEMM: out = act(aA @ Wl + aB @ Wr + bias) ----------------
// All operands pre-packed bf16: inner loop is pure b128 loads + WMMA.
//   A 16x32 bf16 : lane L -> m=L&15, hi=L>>4; two contiguous 8-elem chunks at
//                  k = kb+8*hi and k = kb+16+8*hi  (bf16 row-major -> 2x b128 loads)
//   D 16x16 f32  : vgpr r -> row = r + 8*hi, col = L&15
// ACT: 1 = relu, 2 = sigmoid.  WBF: also write bf16 copy of output (feeds next layer's A).
template<int K, int NCOL, int ACT, bool WBF>
__global__ __launch_bounds__(32 * (NCOL / 16))
void k_sage_gemm(const unsigned short* __restrict__ aA,   // bf16 [N,K], pre-scaled mean agg
                 const unsigned short* __restrict__ aB,   // bf16 [N,K], self features
                 const unsigned short* __restrict__ pWl,
                 const unsigned short* __restrict__ pWr,
                 const float* __restrict__ bias,
                 float* __restrict__ out, unsigned short* __restrict__ outbf, int N) {
  constexpr int CT = NCOL / 16;
  const int lane = threadIdx.x & 31;
  const int wave = threadIdx.x >> 5;
  const int hi   = lane >> 4;
  const int m    = lane & 15;
  const int row  = blockIdx.x * 16 + m;
  const int rowc = (row < N) ? row : (N - 1);   // clamp: keep EXEC all-1s for WMMA
  const int col  = wave * 16 + m;

  const unsigned short* arA = aA + (size_t)rowc * K + 8 * hi;
  const unsigned short* arB = aB + (size_t)rowc * K + 8 * hi;
  const v8u* bl = (const v8u*)(pWl + (size_t)(wave * 32 + lane) * 16);
  const v8u* br = (const v8u*)(pWr + (size_t)(wave * 32 + lane) * 16);

  v8f acc = {};

#pragma unroll
  for (int kb = 0; kb < K; kb += 32) {
    const v8us a0 = *(const v8us*)(arA + kb);
    const v8us a1 = *(const v8us*)(arA + kb + 16);
    const v8us b0 = *(const v8us*)(arB + kb);
    const v8us b1 = *(const v8us*)(arB + kb + 16);
    const v16u ua = __builtin_shufflevector(a0, a1, 0,1,2,3,4,5,6,7,8,9,10,11,12,13,14,15);
    const v16u ub = __builtin_shufflevector(b0, b1, 0,1,2,3,4,5,6,7,8,9,10,11,12,13,14,15);
    const v16bf Bl = __builtin_bit_cast(v16bf, bl[(size_t)(kb >> 5) * CT * 32]);
    const v16bf Br = __builtin_bit_cast(v16bf, br[(size_t)(kb >> 5) * CT * 32]);
    acc = __builtin_amdgcn_wmma_f32_16x16x32_bf16(
        false, __builtin_bit_cast(v16bf, ua), false, Bl, (short)0, acc, false, false);
    acc = __builtin_amdgcn_wmma_f32_16x16x32_bf16(
        false, __builtin_bit_cast(v16bf, ub), false, Br, (short)0, acc, false, false);
  }

  const float bc = bias[col];
#pragma unroll
  for (int r = 0; r < 8; ++r) {
    const int orow = blockIdx.x * 16 + 8 * hi + r;
    if (orow < N) {
      float v = acc[r] + bc;
      if (ACT == 1) v = fmaxf(v, 0.0f);
      if (ACT == 2) v = 1.0f / (1.0f + __expf(-v));
      out[(size_t)orow * NCOL + col] = v;
      if (WBF) outbf[(size_t)orow * NCOL + col] = f2bf(v);
    }
  }
}

// ---------------- launcher ----------------

extern "C" void kernel_launch(void* const* d_in, const int* in_sizes, int n_in,
                              void* d_out, int out_size, void* d_ws, size_t ws_size,
                              hipStream_t stream) {
  const float* x   = (const float*)d_in[0];
  const int*   ei  = (const int*)  d_in[1];
  const float* Wl1 = (const float*)d_in[2];
  const float* Wr1 = (const float*)d_in[3];
  const float* b1  = (const float*)d_in[4];
  const float* Wl2 = (const float*)d_in[5];
  const float* Wr2 = (const float*)d_in[6];
  const float* b2  = (const float*)d_in[7];

  const int N = in_sizes[0] / 64;   // 100000
  const int E = in_sizes[1] / 2;    // 1600000
  const int* src = ei;
  const int* dst = ei + E;

  // workspace: invdeg | h(f32) | agg2(f32) | h_bf | union{x_bf,a1_bf}->a2_bf | packed W
  char* ws = (char*)d_ws;
  size_t off = 0;
  auto carve = [&](size_t bytes) -> char* {
    char* p = ws + off;
    off = (off + bytes + 255) & ~(size_t)255;
    return p;
  };
  float*          invdeg = (float*)carve((size_t)N * 4);
  float*          h      = (float*)carve((size_t)N * 128 * 4);
  float*          agg2   = (float*)carve((size_t)N * 128 * 4);
  unsigned short* h_bf   = (unsigned short*)carve((size_t)N * 128 * 2);
  unsigned short* ubuf   = (unsigned short*)carve((size_t)N * 128 * 2);
  unsigned short* x_bf   = ubuf;                       // live: start .. GEMM1
  unsigned short* a1_bf  = ubuf + (size_t)N * 64;      // live: cvt1  .. GEMM1
  unsigned short* a2_bf  = ubuf;                       // live: cvt2  .. GEMM2
  unsigned short* pWl1 = (unsigned short*)carve((size_t)64 * 128 * 2);
  unsigned short* pWr1 = (unsigned short*)carve((size_t)64 * 128 * 2);
  unsigned short* pWl2 = (unsigned short*)carve((size_t)128 * 64 * 2);
  unsigned short* pWr2 = (unsigned short*)carve((size_t)128 * 64 * 2);
  float* agg1 = (float*)d_out;   // [N,64] f32 scratch; fully overwritten by final GEMM

  hipMemsetAsync(invdeg, 0, (size_t)N * 4,       stream);
  hipMemsetAsync(agg1,   0, (size_t)N * 64 * 4,  stream);
  hipMemsetAsync(agg2,   0, (size_t)N * 128 * 4, stream);

  // pack weights into WMMA fragment order (tiny; once per launch)
  k_pack_w<64, 128><<<(64 * 128 + 255) / 256, 256, 0, stream>>>(Wl1, pWl1);
  k_pack_w<64, 128><<<(64 * 128 + 255) / 256, 256, 0, stream>>>(Wr1, pWr1);
  k_pack_w<128, 64><<<(128 * 64 + 255) / 256, 256, 0, stream>>>(Wl2, pWl2);
  k_pack_w<128, 64><<<(128 * 64 + 255) / 256, 256, 0, stream>>>(Wr2, pWr2);

  // degrees -> reciprocal (mean divisor, isolated nodes -> 1)
  k_degree<<<(E + 255) / 256, 256, 0, stream>>>(dst, invdeg, E);
  k_invdeg<<<(N + 255) / 256, 256, 0, stream>>>(invdeg, N);

  // x -> bf16 (self-features operand of GEMM1)
  {
    long long T = (long long)N * (64 / 4);
    k_cvt_bf<64, false><<<(unsigned)((T + 255) / 256), 256, 0, stream>>>(x, nullptr, x_bf, N);
  }

  // layer 1: scatter x -> agg1; convert (agg1 * invdeg) -> bf16; fused GEMM
  {
    long long T = (long long)E * (64 / 4);
    k_scatter<64><<<(unsigned)((T + 255) / 256), 256, 0, stream>>>(x, src, dst, agg1, E);
  }
  {
    long long T = (long long)N * (64 / 4);
    k_cvt_bf<64, true><<<(unsigned)((T + 255) / 256), 256, 0, stream>>>(agg1, invdeg, a1_bf, N);
  }
  k_sage_gemm<64, 128, 1, true><<<(N + 15) / 16, 256, 0, stream>>>(
      a1_bf, x_bf, pWl1, pWr1, b1, h, h_bf, N);

  // layer 2: scatter h -> agg2; convert (agg2 * invdeg) -> bf16; fused GEMM + sigmoid
  {
    long long T = (long long)E * (128 / 4);
    k_scatter<128><<<(unsigned)((T + 255) / 256), 256, 0, stream>>>(h, src, dst, agg2, E);
  }
  {
    long long T = (long long)N * (128 / 4);
    k_cvt_bf<128, true><<<(unsigned)((T + 255) / 256), 256, 0, stream>>>(agg2, invdeg, a2_bf, N);
  }
  k_sage_gemm<128, 64, 2, false><<<(N + 15) / 16, 128, 0, stream>>>(
      a2_bf, h_bf, pWl2, pWr2, b2, (float*)d_out, nullptr, N);
}
